// MultiTaskHeads_70927089926579
// MI455X (gfx1250) — matrix operand order repaired and checked by reading, not compiled, not run on hardware
//
#include <hip/hip_runtime.h>
#include <hip/hip_bf16.h>

typedef _Float16 v16h __attribute__((ext_vector_type(16)));
typedef _Float16 v8h  __attribute__((ext_vector_type(8)));
typedef float    v8f  __attribute__((ext_vector_type(8)));

#define D_IN   1280
#define H_DIM  512
#define T_HEADS 4
#define KSTEPS 40      // D_IN / 32 (K per WMMA)
#define NBLK   32      // H_DIM / 16 (N blocks per head)
#define WAVES  8
#define ROWS_PER_BLOCK 128   // 8 waves * 16 rows
#define LDS_BYTES (WAVES * KSTEPS * 32 * 16 * 2)   // 327680 B = 320 KB

// ---------------------------------------------------------------------------
// Branchless exact-erf GELU (Abramowitz–Stegun 7.1.26, |err| < 1.5e-7).
// No EXEC divergence (keeps EXEC all-1s between WMMA regions); uses the
// native v_rcp_f32 / v_exp_f32 transcendental pipe which co-executes with XDL.
// ---------------------------------------------------------------------------
__device__ __forceinline__ float gelu_exact(float x) {
    float z = fabsf(x) * 0.70710678118654752f;           // |x|/sqrt(2)
    float t = __builtin_amdgcn_rcpf(fmaf(0.3275911f, z, 1.0f));
    float p = t * fmaf(t, fmaf(t, fmaf(t, fmaf(t, 1.061405429f, -1.453152027f),
                                       1.421413741f), -0.284496736f), 0.254829592f);
    float e = __expf(-z * z);
    float erf_abs = fmaf(-p, e, 1.0f);                   // erf(|x|/sqrt(2))
    float erfv = copysignf(erf_abs, x);
    return 0.5f * x * (1.0f + erfv);
}

// ---------------------------------------------------------------------------
// Pack W1 (T, D, H) f32 row-major  ->  f16 WMMA B-fragments.
// Fragment id = (t*NBLK + nb)*KSTEPS + k ; per fragment 32 lanes x 16 halves.
// B k order used here (contiguous, same as A): lane L<16 holds col N=L, K = k*32 + j (j=0..15);
// lane L>=16 holds col N=L-16, K = k*32 + 16 + j.
// One thread writes one 16B half-slice (8 consecutive K values).
// ---------------------------------------------------------------------------
__global__ __launch_bounds__(256) void pack_w1_kernel(const float* __restrict__ W1,
                                                      _Float16* __restrict__ Bbuf) {
    int u = blockIdx.x * 256 + threadIdx.x;          // 327680 units total
    int h    = u & 1;
    int lane = (u >> 1) & 31;
    int rem  = u >> 6;                               // (t*NBLK + nb)*KSTEPS + k
    int k    = rem % KSTEPS;
    int rem2 = rem / KSTEPS;                         // t*NBLK + nb
    int nb   = rem2 & 31;
    int t    = rem2 >> 5;
    int col    = nb * 16 + (lane & 15);
    int kstart = k * 32 + ((lane < 16) ? 0 : 16) + h * 8;
    v8h hv;
#pragma unroll
    for (int i = 0; i < 8; ++i) {
        float v = W1[((size_t)t * D_IN + (kstart + i)) * H_DIM + col];
        hv[i] = (_Float16)v;
    }
    reinterpret_cast<v8h*>(Bbuf)[u] = hv;
}

// ---------------------------------------------------------------------------
// Fused multi-head kernel: X·W1[t] + b1[t] -> GELU -> ·w2[t] + b2[t] -> route.
// Block = 256 threads (8 waves); each wave owns 16 samples, all 2048 columns.
// ---------------------------------------------------------------------------
__global__ __launch_bounds__(256) void mt_heads_kernel(const float* __restrict__ X,
                                                       const int*   __restrict__ idx,
                                                       const _Float16* __restrict__ Bbuf,
                                                       const float* __restrict__ b1,
                                                       const float* __restrict__ w2,
                                                       const float* __restrict__ b2,
                                                       float* __restrict__ out) {
    extern __shared__ _Float16 ldsA[];   // 8 waves * 40 ksteps * 32 lanes * 16 halves
    const int tid  = threadIdx.x;
    const int lane = tid & 31;
    const int wv   = tid >> 5;
    const int blockBase = blockIdx.x * ROWS_PER_BLOCK;

    // ---- stage features into LDS as f16 WMMA A-fragments ----
    // A k order used here (contiguous, same as B): lane L<16 is row M=L:   j=0..15 -> K=j.
    //                       lane L>=16 is row M=L-16: j=0..15 -> K=j+16.
    // Each 16B half-slice = 8 consecutive K values -> two float4 loads + cvt.
    for (int u = tid; u < WAVES * KSTEPS * 32 * 2; u += 256) {
        int h   = u & 1;
        int ln  = (u >> 1) & 31;
        int rem = u >> 6;                 // w*KSTEPS + k
        int k   = rem % KSTEPS;
        int w   = rem / KSTEPS;
        int row = blockBase + w * 16 + (ln & 15);
        int kstart = k * 32 + ((ln < 16) ? 0 : 16) + h * 8;
        const float4* src = reinterpret_cast<const float4*>(X + (size_t)row * D_IN + kstart);
        float4 f0 = src[0], f1 = src[1];
        v8h hv;
        hv[0] = (_Float16)f0.x; hv[1] = (_Float16)f0.y;
        hv[2] = (_Float16)f0.z; hv[3] = (_Float16)f0.w;
        hv[4] = (_Float16)f1.x; hv[5] = (_Float16)f1.y;
        hv[6] = (_Float16)f1.z; hv[7] = (_Float16)f1.w;
        reinterpret_cast<v8h*>(ldsA)[u] = hv;
    }
    __syncthreads();

    const v16h* Afrag = reinterpret_cast<const v16h*>(ldsA) + (wv * KSTEPS) * 32 + lane;
    const v16h* Bfrag = reinterpret_cast<const v16h*>(Bbuf) + lane;

    // routing indices for this wave's rows (C layout: lanes<16 -> rows 0..7,
    // lanes>=16 -> rows 8..15, VGPR r = row offset)
    const int rowHalf = (lane >> 4) * 8;
    int myidx[8];
#pragma unroll
    for (int r = 0; r < 8; ++r)
        myidx[r] = idx[blockBase + wv * 16 + rowHalf + r];

    float fin[8];
#pragma unroll
    for (int r = 0; r < 8; ++r) fin[r] = 0.0f;

#pragma unroll 1
    for (int t = 0; t < T_HEADS; ++t) {
        float s[8];
#pragma unroll
        for (int r = 0; r < 8; ++r) s[r] = 0.0f;

#pragma unroll 1
        for (int ng = 0; ng < 8; ++ng) {           // 4 n-blocks (64 cols) per pass
            const int nb0 = ng * 4;
            v8f acc0 = {}, acc1 = {}, acc2 = {}, acc3 = {};
#pragma unroll 2
            for (int k = 0; k < KSTEPS; ++k) {
                v16h a   = Afrag[k * 32];
                v16h bv0 = Bfrag[(((t * NBLK + nb0 + 0) * KSTEPS) + k) * 32];
                v16h bv1 = Bfrag[(((t * NBLK + nb0 + 1) * KSTEPS) + k) * 32];
                v16h bv2 = Bfrag[(((t * NBLK + nb0 + 2) * KSTEPS) + k) * 32];
                v16h bv3 = Bfrag[(((t * NBLK + nb0 + 3) * KSTEPS) + k) * 32];
                acc0 = __builtin_amdgcn_wmma_f32_16x16x32_f16(false, a, false, bv0, (short)0, acc0, false, false);
                acc1 = __builtin_amdgcn_wmma_f32_16x16x32_f16(false, a, false, bv1, (short)0, acc1, false, false);
                acc2 = __builtin_amdgcn_wmma_f32_16x16x32_f16(false, a, false, bv2, (short)0, acc2, false, false);
                acc3 = __builtin_amdgcn_wmma_f32_16x16x32_f16(false, a, false, bv3, (short)0, acc3, false, false);
            }
            // fused epilogue: + b1, branchless exact GELU, dot with w2
#pragma unroll
            for (int q = 0; q < 4; ++q) {
                const v8f acc = (q == 0) ? acc0 : (q == 1) ? acc1 : (q == 2) ? acc2 : acc3;
                const int col = (nb0 + q) * 16 + (lane & 15);
                const float bb = b1[t * H_DIM + col];
                const float ww = w2[t * H_DIM + col];
#pragma unroll
                for (int r = 0; r < 8; ++r) {
                    float g = gelu_exact(acc[r] + bb);
                    s[r] = fmaf(g, ww, s[r]);
                }
            }
        }
        // reduce across the 16 lanes of each half (wave32 butterfly)
        const float bias2 = b2[t];
#pragma unroll
        for (int r = 0; r < 8; ++r) {
            float v = s[r];
            v += __shfl_xor(v, 1, 32);
            v += __shfl_xor(v, 2, 32);
            v += __shfl_xor(v, 4, 32);
            v += __shfl_xor(v, 8, 32);
            if (myidx[r] == t) fin[r] = v + bias2;   // route
        }
    }

    if ((lane & 15) == 0) {
#pragma unroll
        for (int r = 0; r < 8; ++r)
            out[blockBase + wv * 16 + rowHalf + r] = fin[r];
    }
}

extern "C" void kernel_launch(void* const* d_in, const int* in_sizes, int n_in,
                              void* d_out, int out_size, void* d_ws, size_t ws_size,
                              hipStream_t stream) {
    const float* features = (const float*)d_in[0];
    const int*   aff_idx  = (const int*)d_in[1];
    const float* W1       = (const float*)d_in[2];
    const float* b1       = (const float*)d_in[3];
    const float* w2       = (const float*)d_in[4];
    const float* b2       = (const float*)d_in[5];
    float*       out      = (float*)d_out;
    _Float16*    Bbuf     = (_Float16*)d_ws;   // 4*32*40*512 halves = 5.25 MB

    // Pack W1 -> f16 B-fragments (327680 16B units / 256 threads = 1280 blocks)
    pack_w1_kernel<<<dim3(1280), dim3(256), 0, stream>>>(W1, Bbuf);

    // 320 KB dynamic LDS per workgroup (full WGP allocation on CDNA5)
    (void)hipFuncSetAttribute(reinterpret_cast<const void*>(mt_heads_kernel),
                              hipFuncAttributeMaxDynamicSharedMemorySize, LDS_BYTES);

    // 65536 samples / 128 per block = 512 blocks of 256 threads (8 waves)
    mt_heads_kernel<<<dim3(512), dim3(256), LDS_BYTES, stream>>>(
        features, aff_idx, Bbuf, b1, w2, b2, out);
}
